// RNN_Hierarchical_28243704939232
// MI455X (gfx1250) — matrix operand order repaired and checked by reading, not compiled, not run on hardware
//
#include <hip/hip_runtime.h>
#include <hip/hip_bf16.h>

// ---------------------------------------------------------------------------
// Hierarchical 3-layer RNN scan for MI455X (gfx1250, wave32, WMMA).
//
//  * 3 workgroups = 3-stage layer pipeline over 1024 timesteps, handing the
//    f16 state through an L2 ring buffer (release/acquire counters).
//  * Per-WG LDS residency: W_hh0 (128K) + h state (32K) + prev-layer stage
//    (32K) + x stage (8K) = 200KB of the 320KB WGP LDS.
//  * W_ff / W_in B-fragments live in VGPRs for the whole scan (per wave only
//    its 16 N-columns: 8+2 v16h), eliminating per-step weight traffic.
//  * Incoming activations staged into LDS once per step (kills the 16x
//    per-wave redundancy on global A-fragment reads), using the CDNA5
//    async global<->LDS DMA path (ASYNCcnt).
//  * GEMMs: v_wmma_f32_16x16x32_f16, K folded as [h(256) | x(64)] = 320.
//
// Workspace layout (needs ~17.6 MB):
//   [0)       ctrs   : 4 x u32 (prod[2], cons[2])
//   [64)      biasA  : 3*256 f32   (b_in + b_hh)
//   [4096)    wh16   : 3*256*256 f16 (W_hh, diag zeroed)
//   [397312)  win16  : 3*256*64  f16
//   [495616)  wf16   : 2*256*256 f16
//   [757760)  data16 : 1024*64*64 f16
//   [9146368) hxfer  : 2 links * RING(128) * 64*256 f16
// ---------------------------------------------------------------------------

typedef __attribute__((ext_vector_type(16))) _Float16 v16h;
typedef __attribute__((ext_vector_type(8)))  _Float16 v8h;
typedef __attribute__((ext_vector_type(8)))  float    v8f;
typedef int v4i_ __attribute__((vector_size(16)));   // builtin's expected pointee

#define T_STEPS 1024
#define RING    128
// LDS: whL(128K) + hbuf(32K) + pinL(32K) + xsL(8K)
#define SMEM_BYTES ((65536 + 16384 + 16384 + 4096) * 2)

#if __has_builtin(__builtin_amdgcn_global_load_async_to_lds_b128) && \
    __has_builtin(__builtin_amdgcn_global_store_async_from_lds_b128) && \
    __has_builtin(__builtin_amdgcn_s_wait_asynccnt)
#define ASYNC_LDS 1
#else
#define ASYNC_LDS 0
#endif

#if ASYNC_LDS
__device__ __forceinline__ void g2l16(_Float16* l, const _Float16* g) {
  __builtin_amdgcn_global_load_async_to_lds_b128(
      (__attribute__((address_space(1))) v4i_*)(g),
      (__attribute__((address_space(3))) v4i_*)(l), 0, 0);
}
__device__ __forceinline__ void l2g16(_Float16* g, const _Float16* l) {
  __builtin_amdgcn_global_store_async_from_lds_b128(
      (__attribute__((address_space(1))) v4i_*)(g),
      (__attribute__((address_space(3))) v4i_*)(l), 0, 0);
}
#endif

__device__ __forceinline__ v16h frag16(const _Float16* p0, const _Float16* p1) {
  v8h lo = *(const v8h*)p0;
  v8h hi = *(const v8h*)p1;
  v16h r;
#pragma unroll
  for (int i = 0; i < 8; ++i) { r[i] = lo[i]; r[i + 8] = hi[i]; }
  return r;
}

// A fragment (16x32 f16), row-major source, row stride ld halves.
// lanes 0-15: row=lane,    K = {k0+0..7, k0+16..23}
// lanes16-31: row=lane-16, K = {k0+8..15, k0+24..31}
__device__ __forceinline__ v16h loadA(const _Float16* tb, int ld, int lane, int k0) {
  const int row = lane & 15;
  const int s0  = (lane & 16) ? 8 : 0;
  const _Float16* p = tb + row * ld + k0 + s0;
  return frag16(p, p + 16);
}

// B fragment (32x16 f16) from row-major W[n][k] (pre = h @ W^T).
// lanes 0-15: n=n0+lane, K={k0+0..7, k0+16..23} ; lanes 16-31: K={k0+8..15, k0+24..31}
__device__ __forceinline__ v16h loadB(const _Float16* wb, int ld, int lane, int n0, int k0) {
  const int col = n0 + (lane & 15);
  const int s0  = (lane & 16) ? 8 : 0;
  const _Float16* p = wb + col * ld + k0 + s0;
  return frag16(p, p + 16);
}

__device__ __forceinline__ v8f wmma16(v16h a, v16h b, v8f c) {
  return __builtin_amdgcn_wmma_f32_16x16x32_f16(false, a, false, b, (short)0, c, false, false);
}

__device__ __forceinline__ float lrelu(float z) { return z >= 0.f ? z : 0.01f * z; }

// ---------------------------------------------------------------------------
__global__ __launch_bounds__(512) void rnn_prep(
    const float* __restrict__ data, const float* __restrict__ W_in,
    const float* __restrict__ b_in, const float* __restrict__ W_hh,
    const float* __restrict__ b_hh, const float* __restrict__ W_ff,
    _Float16* __restrict__ data16, _Float16* __restrict__ wh16,
    _Float16* __restrict__ win16, _Float16* __restrict__ wf16,
    float* __restrict__ biasA, unsigned* __restrict__ ctrs) {
  const int tid = blockIdx.x * blockDim.x + threadIdx.x;
  const int stride = gridDim.x * blockDim.x;
  if (tid < 4) ctrs[tid] = 0u;
  for (int i = tid; i < T_STEPS * 64 * 64; i += stride) data16[i] = (_Float16)data[i];
  for (int i = tid; i < 3 * 256 * 256; i += stride) {
    const int k = i & 255, n = (i >> 8) & 255;
    wh16[i] = (n == k) ? (_Float16)0.f : (_Float16)W_hh[i];   // zero recurrent diagonal
  }
  for (int i = tid; i < 3 * 256 * 64; i += stride) win16[i] = (_Float16)W_in[i];
  for (int i = tid; i < 2 * 256 * 256; i += stride) wf16[i] = (_Float16)W_ff[i];
  for (int i = tid; i < 3 * 256; i += stride) biasA[i] = b_in[i] + b_hh[i];
}

// ---------------------------------------------------------------------------
__global__ __launch_bounds__(512, 1) void rnn_scan(
    const float* __restrict__ h0, const float* __restrict__ b_ff,
    const float* __restrict__ taus, const float* __restrict__ W_fc,
    const float* __restrict__ b_fc, const _Float16* __restrict__ data16,
    const _Float16* __restrict__ wh16, const _Float16* __restrict__ win16,
    const _Float16* __restrict__ wf16, const float* __restrict__ biasA,
    _Float16* __restrict__ hxfer, unsigned* __restrict__ ctrs,
    float* __restrict__ out) {
  extern __shared__ char smem[];
  _Float16* whL  = (_Float16*)smem;        // [256][256] own-layer W_hh0
  _Float16* hbuf = whL + 65536;            // [64][256]  current h (f16)
  _Float16* pinL = hbuf + 16384;           // [64][256]  staged new[d-1](t)
  _Float16* xsL  = pinL + 16384;           // [64][64]   staged data16[t]

  const int d    = blockIdx.x;             // layer == pipeline stage
  const int tid  = threadIdx.x;
  const int lane = tid & 31;
  const int w    = tid >> 5;               // wave id 0..15 -> N-tile

  // ---- stage resident W_hh0 into LDS, init h state ------------------------
  {
    const _Float16* src = wh16 + (size_t)d * 65536;
#if ASYNC_LDS
    for (int i = tid; i < 8192; i += 512) g2l16(whL + i * 8, src + i * 8);
    __builtin_amdgcn_s_wait_asynccnt(0);
#else
    const uint4* s = (const uint4*)src;
    uint4* dst = (uint4*)whL;
    for (int i = tid; i < 8192; i += 512) dst[i] = s[i];
#endif
    for (int i = tid; i < 64 * 256; i += 512)
      hbuf[i] = (_Float16)h0[(size_t)d * 16384 + i];
  }

  // ---- per-lane constants (C/D layout: N = w*16 + lane%16) ----------------
  const int n = w * 16 + (lane & 15);
  const float bA = biasA[d * 256 + n];
  const float bF = (d > 0) ? b_ff[(d - 1) * 256 + n] : 0.f;
  float tc = taus[d * 256 + n]; if (tc < 1.f) tc = 1.f;
  const float invtau = 1.f / tc;
  const float decay  = 1.f - invtau;

  // ---- per-wave weight fragments held in VGPRs for the whole scan ---------
  v16h wf_reg[8];                                // W_ff[d-1], this wave's 16 cols
  {
    const _Float16* wfsrc = wf16 + (size_t)(d > 0 ? d - 1 : 0) * 65536;
#pragma unroll
    for (int kc = 0; kc < 8; ++kc) wf_reg[kc] = loadB(wfsrc, 256, lane, w * 16, kc * 32);
  }
  v16h win_reg[2];                               // W_in[d], this wave's 16 cols
#pragma unroll
  for (int kx = 0; kx < 2; ++kx)
    win_reg[kx] = loadB(win16 + (size_t)d * 16384, 64, lane, w * 16, kx * 32);

  // ---- f32 state kept in accumulator-fragment layout in VGPRs -------------
  v8f hfrag[4];
#pragma unroll
  for (int mt = 0; mt < 4; ++mt)
#pragma unroll
    for (int r = 0; r < 8; ++r) {
      const int m = mt * 16 + ((lane & 16) ? 8 + r : r);
      hfrag[mt][r] = h0[(size_t)d * 16384 + m * 256 + n];
    }
  __syncthreads();

  unsigned* prod = ctrs;       // prod[l]: steps published on link l
  unsigned* cons = ctrs + 2;   // cons[l]: steps consumed on link l
  _Float16* myout = hxfer + (size_t)d * RING * 16384;                       // d<2
  const _Float16* insrc = (d > 0) ? hxfer + (size_t)(d - 1) * RING * 16384 : nullptr;

  for (int t = 0; t < T_STEPS; ++t) {
    if (t + 1 < T_STEPS)   // L2 prefetch of next input slice
      __builtin_prefetch(data16 + (size_t)(t + 1) * 4096 + tid * 8, 0, 1);

    // ---- pipeline waits (one scalar thread spins; barrier broadcasts) ----
    if (tid == 0) {
      if (d < 2 && t >= RING) {                     // ring back-pressure
        while (__hip_atomic_load(&cons[d], __ATOMIC_RELAXED, __HIP_MEMORY_SCOPE_AGENT)
               < (unsigned)(t - RING + 1))
          __builtin_amdgcn_s_sleep(2);
      }
      if (d > 0) {                                  // wait for producer
        while (__hip_atomic_load(&prod[d - 1], __ATOMIC_RELAXED, __HIP_MEMORY_SCOPE_AGENT)
               < (unsigned)(t + 1))
          __builtin_amdgcn_s_sleep(2);
      }
    }
    __syncthreads();
    if (d > 0) __builtin_amdgcn_fence(__ATOMIC_ACQUIRE, "agent");  // fresh hxfer

    // ---- stage this step's activations into LDS (DMA overlaps WMMAs) -----
    const _Float16* xrow = data16 + (size_t)t * 4096;     // [64][64] f16
    const _Float16* pin  = (d > 0) ? insrc + (size_t)(t & (RING - 1)) * 16384 : nullptr;
#if ASYNC_LDS
    g2l16(xsL + tid * 8, xrow + tid * 8);                             // 8KB
    if (d > 0)
      for (int i = tid; i < 2048; i += 512) g2l16(pinL + i * 8, pin + i * 8);  // 32KB
#else
    ((uint4*)xsL)[tid] = ((const uint4*)xrow)[tid];
    if (d > 0)
      for (int i = tid; i < 2048; i += 512)
        ((uint4*)pinL)[i] = ((const uint4*)pin)[i];
#endif

    // ---- phase A (recurrent K-chunks): pre += h @ W_hh0^T  ---------------
    v8f accA[4];
#pragma unroll
    for (int mt = 0; mt < 4; ++mt)
#pragma unroll
      for (int i = 0; i < 8; ++i) accA[mt][i] = 0.f;

#pragma unroll
    for (int kc = 0; kc < 8; ++kc) {
      const v16h bf = loadB(whL, 256, lane, w * 16, kc * 32);
#pragma unroll
      for (int mt = 0; mt < 4; ++mt) {
        const v16h af = loadA(hbuf + mt * 16 * 256, 256, lane, kc * 32);
        accA[mt] = wmma16(af, bf, accA[mt]);
      }
    }

#if ASYNC_LDS
    __builtin_amdgcn_s_wait_asynccnt(0);
#endif
    __syncthreads();   // staged xsL/pinL visible; hbuf reads complete

    // ---- phase A (input K-chunks): pre += x_t @ W_in^T  (B in VGPRs) -----
#pragma unroll
    for (int kx = 0; kx < 2; ++kx)
#pragma unroll
      for (int mt = 0; mt < 4; ++mt) {
        const v16h af = loadA(xsL + mt * 16 * 64, 64, lane, kx * 32);
        accA[mt] = wmma16(af, win_reg[kx], accA[mt]);
      }

    // ---- phase F: ff = new[d-1](t) @ W_ff^T  (B in VGPRs) ----------------
    v8f accF[4];
#pragma unroll
    for (int mt = 0; mt < 4; ++mt)
#pragma unroll
      for (int i = 0; i < 8; ++i) accF[mt][i] = 0.f;
    if (d > 0) {
#pragma unroll
      for (int kc = 0; kc < 8; ++kc)
#pragma unroll
        for (int mt = 0; mt < 4; ++mt) {
          const v16h af = loadA(pinL + mt * 16 * 256, 256, lane, kc * 32);
          accF[mt] = wmma16(af, wf_reg[kc], accF[mt]);
        }
    }

    // ---- elementwise update (f32 state), write hbuf ----------------------
    _Float16* gout = (d < 2) ? myout + (size_t)(t & (RING - 1)) * 16384 : nullptr;
#pragma unroll
    for (int mt = 0; mt < 4; ++mt) {
#pragma unroll
      for (int r = 0; r < 8; ++r) {
        float pre = accA[mt][r] + bA;
        if (d > 0) pre += lrelu(accF[mt][r] + bF);
        const float nh = decay * hfrag[mt][r] + lrelu(pre) * invtau;
        hfrag[mt][r] = nh;
        const int m = mt * 16 + ((lane & 16) ? 8 + r : r);
        const _Float16 hv = (_Float16)nh;
        hbuf[m * 256 + n] = hv;                 // intra-layer state for t+1
#if !ASYNC_LDS
        if (d < 2) gout[m * 256 + n] = hv;      // handoff (fallback path)
#endif
      }
    }
    __syncthreads();   // hbuf fully updated; pinL/xsL reads complete

#if ASYNC_LDS
    if (d < 2) {       // bulk LDS -> L2 handoff via async DMA
      for (int i = tid; i < 2048; i += 512) l2g16(gout + i * 8, hbuf + i * 8);
      __builtin_amdgcn_s_wait_asynccnt(0);
    }
#endif
    __threadfence();
    __syncthreads();
    if (tid == 0) {
      if (d < 2)
        __hip_atomic_store(&prod[d], (unsigned)(t + 1), __ATOMIC_RELEASE, __HIP_MEMORY_SCOPE_AGENT);
      if (d > 0)
        __hip_atomic_store(&cons[d - 1], (unsigned)(t + 1), __ATOMIC_RELEASE, __HIP_MEMORY_SCOPE_AGENT);
    }
  }

  // ---- hs_final: f32 fragments -> d_out[d][b][h] --------------------------
  float* hs_out = out + (size_t)d * 16384;
#pragma unroll
  for (int mt = 0; mt < 4; ++mt)
#pragma unroll
    for (int r = 0; r < 8; ++r) {
      const int m = mt * 16 + ((lane & 16) ? 8 + r : r);
      hs_out[m * 256 + n] = hfrag[mt][r];
    }

  // ---- fc heads (all read top layer state): WG2 only, tiny ----------------
  if (d == 2) {
    __syncthreads();
    for (int o = tid; o < 3 * 64 * 2; o += 512) {
      const int dd = o >> 7, rem = o & 127, b = rem >> 1, c = rem & 1;
      const float* wrow = W_fc + (size_t)(dd * 2 + c) * 256;
      float s = b_fc[dd * 2 + c];
      for (int k = 0; k < 256; ++k) s += (float)hbuf[b * 256 + k] * wrow[k];
      out[3 * 64 * 256 + o] = s;
    }
  }
}

// ---------------------------------------------------------------------------
extern "C" void kernel_launch(void* const* d_in, const int* in_sizes, int n_in,
                              void* d_out, int out_size, void* d_ws, size_t ws_size,
                              hipStream_t stream) {
  const float* data = (const float*)d_in[0];
  const float* h0   = (const float*)d_in[1];
  const float* W_in = (const float*)d_in[2];
  const float* b_in = (const float*)d_in[3];
  const float* W_hh = (const float*)d_in[4];
  const float* b_hh = (const float*)d_in[5];
  const float* W_ff = (const float*)d_in[6];
  const float* b_ff = (const float*)d_in[7];
  const float* taus = (const float*)d_in[8];
  const float* W_fc = (const float*)d_in[9];
  const float* b_fc = (const float*)d_in[10];
  float* out = (float*)d_out;

  char* ws = (char*)d_ws;
  unsigned* ctrs    = (unsigned*)ws;
  float*    biasA   = (float*)(ws + 64);
  _Float16* wh16    = (_Float16*)(ws + 4096);
  _Float16* win16   = (_Float16*)(ws + 397312);
  _Float16* wf16    = (_Float16*)(ws + 495616);
  _Float16* data16  = (_Float16*)(ws + 757760);
  _Float16* hxfer   = (_Float16*)(ws + 9146368);   // needs ws >= ~17.6 MB

  (void)in_sizes; (void)n_in; (void)out_size; (void)ws_size;
  (void)hipFuncSetAttribute((const void*)rnn_scan,
                            hipFuncAttributeMaxDynamicSharedMemorySize, SMEM_BYTES);

  rnn_prep<<<1024, 512, 0, stream>>>(data, W_in, b_in, W_hh, b_hh, W_ff,
                                     data16, wh16, win16, wf16, biasA, ctrs);
  rnn_scan<<<3, 512, SMEM_BYTES, stream>>>(h0, b_ff, taus, W_fc, b_fc, data16,
                                           wh16, win16, wf16, biasA, hxfer,
                                           ctrs, out);
}
